// DGCNNModel_21955872817623
// MI455X (gfx1250) — compile-verified, matrix operand order checked by reading, not run on hardware
//
#include <hip/hip_runtime.h>

// ---------------- constants from the reference ----------------
constexpr int cN   = 65536;
constexpr int cE   = 1048576;
constexpr int cB   = 512;
constexpr int cNPG = 128;
constexpr int cIN  = 128;
constexpr int cH   = 256;
constexpr int cK   = 64;     // sort-pool k
constexpr int cTLD = 769;
constexpr int cOUT = 10;

// ---------------- bf16 helpers ----------------
typedef __attribute__((ext_vector_type(16))) __bf16 v16bf;
typedef __attribute__((ext_vector_type(8)))  float  v8f;

__device__ __forceinline__ unsigned short f2bf(float f) {
    unsigned u = __float_as_uint(f);
    u += 0x7FFFu + ((u >> 16) & 1u);          // round-to-nearest-even
    return (unsigned short)(u >> 16);
}
__device__ __forceinline__ float bf2f(unsigned short h) {
    return __uint_as_float(((unsigned)h) << 16);
}

union Frag { v16bf v; uint4 q[2]; };

// ---------------- degree / norm ----------------
__global__ void k_deg_init(float* deg) {
    int i = blockIdx.x * blockDim.x + threadIdx.x;
    if (i < cN) deg[i] = 1.0f;                // self loop
}
__global__ void k_deg_acc(const int* __restrict__ src, float* deg) {
    int e = blockIdx.x * blockDim.x + threadIdx.x;
    if (e < cE) atomicAdd(&deg[src[e]], 1.0f);
}
__global__ void k_inv(const float* __restrict__ deg, float* inv) {
    int i = blockIdx.x * blockDim.x + threadIdx.x;
    if (i < cN) inv[i] = 1.0f / deg[i];
}

// ---------------- fp32 -> bf16 converters ----------------
__global__ void k_cvt_x(const float* __restrict__ x, unsigned short* __restrict__ o, int total4) {
    int i = blockIdx.x * blockDim.x + threadIdx.x;
    if (i >= total4) return;
    int base = i * 4;
    float4 v = *(const float4*)(x + base);
    ushort4 r = make_ushort4(f2bf(v.x), f2bf(v.y), f2bf(v.z), f2bf(v.w));
    *(ushort4*)(o + base) = r;
}
// Wt[n*K + k] = bf16(W[k*Nout + n])   (transpose + convert)
__global__ void k_cvt_w(const float* __restrict__ W, unsigned short* __restrict__ Wt, int Kd, int Nout) {
    int i = blockIdx.x * blockDim.x + threadIdx.x;
    if (i >= Kd * Nout) return;
    int n = i / Kd, k = i - n * Kd;
    Wt[i] = f2bf(W[(size_t)k * Nout + n]);
}

// ---------------- WMMA GEMM: C[M,Nout] = A(bf16 row-major, ld=KD) @ Wt(bf16 [Nout,KD]) + bias ----------------
template <int KD>
__global__ __launch_bounds__(256) void k_gemm(const unsigned short* __restrict__ A,
                                              const unsigned short* __restrict__ Bt,
                                              const float* __restrict__ bias,
                                              float* __restrict__ C, int M, int Nout) {
    const int lane = threadIdx.x & 31;
    const int wave = blockIdx.x * (blockDim.x >> 5) + (threadIdx.x >> 5);
    const int colGroups = Nout >> 6;                 // 64 cols per wave
    const int mt = wave / colGroups;
    const int cg = wave - mt * colGroups;
    if (mt * 16 >= M) return;
    const int l15  = lane & 15;
    const int koff = (lane >> 4) << 3;               // 0 or 8 (ISA 16-bit A/B lane layout)
    const unsigned short* Ar = A  + (size_t)(mt * 16 + l15) * KD + koff;
    const unsigned short* B0 = Bt + (size_t)(cg * 64 + l15) * KD + koff;
    const unsigned short* B1 = B0 + (size_t)16 * KD;
    const unsigned short* B2 = B0 + (size_t)32 * KD;
    const unsigned short* B3 = B0 + (size_t)48 * KD;
    v8f c0 = {}, c1 = {}, c2 = {}, c3 = {};
#pragma unroll
    for (int k = 0; k < KD; k += 32) {
        Frag a, f0, f1, f2, f3;
        a.q[0]  = *(const uint4*)(Ar + k);  a.q[1]  = *(const uint4*)(Ar + k + 16);
        f0.q[0] = *(const uint4*)(B0 + k);  f0.q[1] = *(const uint4*)(B0 + k + 16);
        f1.q[0] = *(const uint4*)(B1 + k);  f1.q[1] = *(const uint4*)(B1 + k + 16);
        f2.q[0] = *(const uint4*)(B2 + k);  f2.q[1] = *(const uint4*)(B2 + k + 16);
        f3.q[0] = *(const uint4*)(B3 + k);  f3.q[1] = *(const uint4*)(B3 + k + 16);
        c0 = __builtin_amdgcn_wmma_f32_16x16x32_bf16(false, a.v, false, f0.v, (short)0, c0, false, false);
        c1 = __builtin_amdgcn_wmma_f32_16x16x32_bf16(false, a.v, false, f1.v, (short)0, c1, false, false);
        c2 = __builtin_amdgcn_wmma_f32_16x16x32_bf16(false, a.v, false, f2.v, (short)0, c2, false, false);
        c3 = __builtin_amdgcn_wmma_f32_16x16x32_bf16(false, a.v, false, f3.v, (short)0, c3, false, false);
    }
    const int rbase = mt * 16 + ((lane >> 4) << 3);  // C layout: lane = N + 16*(M>=8), vgpr r = M&7
    const int cx = cg * 64 + l15;
    const float bi0 = bias[cx], bi1 = bias[cx + 16], bi2 = bias[cx + 32], bi3 = bias[cx + 48];
#pragma unroll
    for (int r = 0; r < 8; ++r) {
        float* Cr = C + (size_t)(rbase + r) * Nout + cx;
        Cr[0]  = c0[r] + bi0;
        Cr[16] = c1[r] + bi1;
        Cr[32] = c2[r] + bi2;
        Cr[48] = c3[r] + bi3;
    }
}

// ---------------- aggregation ----------------
__global__ void k_copy4(const float* __restrict__ s, float* __restrict__ d, int n4) {
    int i = blockIdx.x * blockDim.x + threadIdx.x;
    if (i < n4) ((float4*)d)[i] = ((const float4*)s)[i];
}
// acc[dst] += hL[src], 4 channels per thread
__global__ void k_scatter(const int* __restrict__ src, const int* __restrict__ dst,
                          const float* __restrict__ hL, float* __restrict__ acc) {
    int gid = blockIdx.x * blockDim.x + threadIdx.x;   // E*64 threads
    int e = gid >> 6;
    int c = (gid & 63) << 2;
    float4 v = *(const float4*)(hL + (size_t)src[e] * cH + c);
    float* p = acc + (size_t)dst[e] * cH + c;
    atomicAdd(p + 0, v.x);
    atomicAdd(p + 1, v.y);
    atomicAdd(p + 2, v.z);
    atomicAdd(p + 3, v.w);
}
// out_bf16 = bf16(tanh(acc * inv_deg[node]))
__global__ void k_finalize(const float* __restrict__ acc, const float* __restrict__ inv,
                           unsigned short* __restrict__ o, int total4) {
    int i = blockIdx.x * blockDim.x + threadIdx.x;
    if (i >= total4) return;
    int base = i * 4;
    float s = inv[base >> 8];                  // cH == 256
    float4 v = *(const float4*)(acc + base);
    ushort4 r = make_ushort4(f2bf(tanhf(v.x * s)), f2bf(tanhf(v.y * s)),
                             f2bf(tanhf(v.z * s)), f2bf(tanhf(v.w * s)));
    *(ushort4*)(o + base) = r;
}

// ---------------- layer 4: [N,256] @ [256,1] + b, one wave per node ----------------
__global__ __launch_bounds__(256) void k_layer4(const unsigned short* __restrict__ hA,
                                                const float* __restrict__ W4,
                                                const float* __restrict__ b4,
                                                float* __restrict__ lin4) {
    int wave = blockIdx.x * (blockDim.x >> 5) + (threadIdx.x >> 5);
    int lane = threadIdx.x & 31;
    if (wave >= cN) return;
    const unsigned short* h = hA + (size_t)wave * cH + lane * 8;
    const float* w = W4 + lane * 8;
    float s = 0.0f;
#pragma unroll
    for (int j = 0; j < 8; ++j) s += bf2f(h[j]) * w[j];
#pragma unroll
    for (int off = 16; off; off >>= 1) s += __shfl_xor(s, off, 32);
    if (lane == 0) lin4[wave] = s + b4[0];
}
__global__ void k_copy1(const float* __restrict__ s, float* __restrict__ d) {
    int i = blockIdx.x * blockDim.x + threadIdx.x;
    if (i < cN) d[i] = s[i];
}
__global__ void k_scatter1(const int* __restrict__ src, const int* __restrict__ dst,
                           const float* __restrict__ lin4, float* __restrict__ acc) {
    int e = blockIdx.x * blockDim.x + threadIdx.x;
    if (e < cE) atomicAdd(&acc[dst[e]], lin4[src[e]]);
}
__global__ void k_fin1(const float* __restrict__ acc, const float* __restrict__ inv,
                       float* __restrict__ score) {
    int i = blockIdx.x * blockDim.x + threadIdx.x;
    if (i < cN) score[i] = tanhf(acc[i] * inv[i]);
}

// ---------------- sort pool: descending bitonic sort of 128, keep top 64 ----------------
__global__ __launch_bounds__(128) void k_sortpool(const float* __restrict__ score, float* __restrict__ top) {
    __shared__ float s[cNPG];
    int b = blockIdx.x, t = threadIdx.x;
    s[t] = score[b * cNPG + t];
    __syncthreads();
    for (int k = 2; k <= cNPG; k <<= 1)
        for (int j = k >> 1; j > 0; j >>= 1) {
            int ixj = t ^ j;
            if (ixj > t) {
                float a = s[t], c = s[ixj];
                bool up = ((t & k) == 0);      // descending final order
                if ((a < c) == up) { s[t] = c; s[ixj] = a; }
            }
            __syncthreads();
        }
    if (t < cK) top[b * cK + t] = s[t];
}

// ---------------- tail: conv1(stride=kernel=TLD) -> relu -> maxpool2 -> conv2 -> relu -> fc1 -> relu -> fc2 ----------------
__global__ __launch_bounds__(128) void k_tail(const float* __restrict__ top,
                                              const float* __restrict__ c1w, const float* __restrict__ c1b,
                                              const float* __restrict__ c2w, const float* __restrict__ c2b,
                                              const float* __restrict__ f1w, const float* __restrict__ f1b,
                                              const float* __restrict__ f2w, const float* __restrict__ f2b,
                                              float* __restrict__ out) {
    __shared__ float sTop[cK];
    __shared__ float sYd[16];
    __shared__ float sP[16][32];
    __shared__ float sFlat[896];
    __shared__ float sF1[128];
    int b = blockIdx.x, t = threadIdx.x;
    if (t < cK) sTop[t] = top[b * cK + t];
    __syncthreads();
    // conv1 output position 0 is the only data-dependent one (kernel=stride=TLD, zero pad)
    if (t < 16) {
        const float* w = c1w + t * cTLD;
        float d = c1b[t];
        for (int j = 0; j < cK; ++j) d += sTop[j] * w[j];
        sYd[t] = d;
    }
    __syncthreads();
    // relu + maxpool(2): positions >=1 of conv1 are pure bias
    for (int idx = t; idx < 16 * 32; idx += 128) {
        int c = idx >> 5, p = idx & 31;
        float br = fmaxf(c1b[c], 0.0f);
        sP[c][p] = (p == 0) ? fmaxf(fmaxf(sYd[c], 0.0f), br) : br;
    }
    __syncthreads();
    // conv2: [16,32] -> [32,28], kernel 5, relu, flatten o*28+t
    for (int idx = t; idx < 896; idx += 128) {
        int o = idx / 28, tt = idx - o * 28;
        float acc = c2b[o];
        const float* w = c2w + o * 80;
#pragma unroll
        for (int i = 0; i < 16; ++i)
#pragma unroll
            for (int k = 0; k < 5; ++k)
                acc += sP[i][tt + k] * w[i * 5 + k];
        sFlat[idx] = fmaxf(acc, 0.0f);
    }
    __syncthreads();
    // fc1: 896 -> 128, relu  (coalesced: thread t reads f1w[v*128 + t])
    {
        float acc = f1b[t];
        for (int v = 0; v < 896; ++v) acc += sFlat[v] * f1w[v * 128 + t];
        sF1[t] = fmaxf(acc, 0.0f);
    }
    __syncthreads();
    // fc2: 128 -> 10
    if (t < cOUT) {
        float acc = f2b[t];
#pragma unroll
        for (int u = 0; u < 128; ++u) acc += sF1[u] * f2w[u * cOUT + t];
        out[b * cOUT + t] = acc;
    }
}

// ---------------- launch ----------------
extern "C" void kernel_launch(void* const* d_in, const int* in_sizes, int n_in,
                              void* d_out, int out_size, void* d_ws, size_t ws_size,
                              hipStream_t stream) {
    const float* x   = (const float*)d_in[0];
    const int*   ei  = (const int*)d_in[1];
    const int*   src = ei;
    const int*   dst = ei + cE;
    const float* W1 = (const float*)d_in[3];  const float* b1 = (const float*)d_in[4];
    const float* W2 = (const float*)d_in[5];  const float* b2 = (const float*)d_in[6];
    const float* W3 = (const float*)d_in[7];  const float* b3 = (const float*)d_in[8];
    const float* W4 = (const float*)d_in[9];  const float* b4 = (const float*)d_in[10];
    const float* c1w = (const float*)d_in[11]; const float* c1b = (const float*)d_in[12];
    const float* c2w = (const float*)d_in[13]; const float* c2b = (const float*)d_in[14];
    const float* f1w = (const float*)d_in[15]; const float* f1b = (const float*)d_in[16];
    const float* f2w = (const float*)d_in[17]; const float* f2b = (const float*)d_in[18];

    char* ws = (char*)d_ws;
    size_t off = 0;
    auto carve = [&](size_t bytes) -> void* {
        void* p = ws + off;
        off = (off + bytes + 255) & ~(size_t)255;
        return p;
    };
    unsigned short* hA   = (unsigned short*)carve((size_t)cN * cH * 2);
    float*          hL   = (float*)carve((size_t)cN * cH * 4);
    float*          hAcc = (float*)carve((size_t)cN * cH * 4);
    float*          deg  = (float*)carve((size_t)cN * 4);
    float*          invd = (float*)carve((size_t)cN * 4);
    unsigned short* wt1  = (unsigned short*)carve((size_t)cIN * cH * 2);
    unsigned short* wt2  = (unsigned short*)carve((size_t)cH * cH * 2);
    unsigned short* wt3  = (unsigned short*)carve((size_t)cH * cH * 2);
    float*          lin4 = (float*)carve((size_t)cN * 4);
    float*          acc1 = (float*)carve((size_t)cN * 4);
    float*          scr  = (float*)carve((size_t)cN * 4);
    float*          top  = (float*)carve((size_t)cB * cK * 4);
    (void)ws_size; (void)in_sizes; (void)n_in; (void)out_size;

    const int T = 256;
    // degree / norm
    k_deg_init<<<(cN + T - 1) / T, T, 0, stream>>>(deg);
    k_deg_acc<<<(cE + T - 1) / T, T, 0, stream>>>(src, deg);
    k_inv<<<(cN + T - 1) / T, T, 0, stream>>>(deg, invd);
    // convert inputs / weights to bf16
    k_cvt_x<<<((cN * cIN / 4) + T - 1) / T, T, 0, stream>>>(x, hA, cN * cIN / 4);
    k_cvt_w<<<((cIN * cH) + T - 1) / T, T, 0, stream>>>(W1, wt1, cIN, cH);
    k_cvt_w<<<((cH * cH) + T - 1) / T, T, 0, stream>>>(W2, wt2, cH, cH);
    k_cvt_w<<<((cH * cH) + T - 1) / T, T, 0, stream>>>(W3, wt3, cH, cH);

    const int gemmBlocks = (cN / 16) * (cH / 64) / 8;   // 8 waves per 256-thread block
    const int n4 = cN * cH / 4;
    const int scatBlocks = (cE * 64) / T;

    // layer 1
    k_gemm<cIN><<<gemmBlocks, T, 0, stream>>>(hA, wt1, b1, hL, cN, cH);
    k_copy4<<<(n4 + T - 1) / T, T, 0, stream>>>(hL, hAcc, n4);
    k_scatter<<<scatBlocks, T, 0, stream>>>(src, dst, hL, hAcc);
    k_finalize<<<(n4 + T - 1) / T, T, 0, stream>>>(hAcc, invd, hA, n4);
    // layer 2
    k_gemm<cH><<<gemmBlocks, T, 0, stream>>>(hA, wt2, b2, hL, cN, cH);
    k_copy4<<<(n4 + T - 1) / T, T, 0, stream>>>(hL, hAcc, n4);
    k_scatter<<<scatBlocks, T, 0, stream>>>(src, dst, hL, hAcc);
    k_finalize<<<(n4 + T - 1) / T, T, 0, stream>>>(hAcc, invd, hA, n4);
    // layer 3
    k_gemm<cH><<<gemmBlocks, T, 0, stream>>>(hA, wt3, b3, hL, cN, cH);
    k_copy4<<<(n4 + T - 1) / T, T, 0, stream>>>(hL, hAcc, n4);
    k_scatter<<<scatBlocks, T, 0, stream>>>(src, dst, hL, hAcc);
    k_finalize<<<(n4 + T - 1) / T, T, 0, stream>>>(hAcc, invd, hA, n4);
    // layer 4 (scalar channel)
    k_layer4<<<(cN * 32) / T, T, 0, stream>>>(hA, W4, b4, lin4);
    k_copy1<<<(cN + T - 1) / T, T, 0, stream>>>(lin4, acc1);
    k_scatter1<<<(cE + T - 1) / T, T, 0, stream>>>(src, dst, lin4, acc1);
    k_fin1<<<(cN + T - 1) / T, T, 0, stream>>>(acc1, invd, scr);
    // sort-pool + CNN/MLP tail
    k_sortpool<<<cB, cNPG, 0, stream>>>(scr, top);
    k_tail<<<cB, 128, 0, stream>>>(top, c1w, c1b, c2w, c2b, f1w, f1b, f2w, f2b, (float*)d_out);
}
